// EdgeNet_1675037245678
// MI455X (gfx1250) — matrix operand order repaired and checked by reading, not compiled, not run on hardware
//
#include <hip/hip_runtime.h>

// ---------------------------------------------------------------------------
// EdgeNet on MI455X (gfx1250): bf16-preconverted x -> async global->LDS gather
// -> [64,288]bf16 per block -> 3x GEMM via v_wmma_f32_16x16x32_bf16 (f32 acc),
// ELU fused, bf16 inter-layer LDS ping-pong.
// ---------------------------------------------------------------------------

typedef __bf16 v8bf  __attribute__((ext_vector_type(8)));
typedef __bf16 v16bf __attribute__((ext_vector_type(16)));
typedef float  v8f   __attribute__((ext_vector_type(8)));

union V16BF { v16bf v; v8bf h[2]; };

#define D_FEAT 128
#define HID    256
#define OUTC   128
#define KPAD0  288   // 259 input channels permuted+padded to 9*32
#define LDH    264   // LDS stride for 256-wide activations (16B-aligned)
#define MBLK   64    // edges per block (4 M-tiles of 16)
#define MT     4     // M-tiles per block

#if __has_builtin(__builtin_amdgcn_global_load_async_to_lds_b128)
#define HAVE_ASYNC_LDS 1
// Param types per hipcc diagnostic: global (addrspace(1)) pointer to a
// 16-byte int vector; LDS side is the addrspace(3) counterpart.
typedef int v4i __attribute__((vector_size(16)));
typedef __attribute__((address_space(1))) v4i Gv4i;
typedef __attribute__((address_space(3))) v4i Lv4i;
#else
#define HAVE_ASYNC_LDS 0
#endif

// ---------------------------------------------------------------------------
// Weight prep: fp32 -> bf16, W0 columns permuted to K' = [x_src | x_dst | ea | 0pad]
// ---------------------------------------------------------------------------
__global__ __launch_bounds__(256) void edgenet_prep(
    const float* __restrict__ W0, const float* __restrict__ W2,
    const float* __restrict__ W3,
    __bf16* __restrict__ W0b, __bf16* __restrict__ W2b, __bf16* __restrict__ W3b) {
  const int t = blockIdx.x * 256 + threadIdx.x;
  const int N0 = HID * KPAD0;       // 73728
  const int N2 = HID * HID;         // 65536
  const int N3 = OUTC * HID;        // 32768
  if (t < N0) {
    const int n = t / KPAD0, j = t % KPAD0;
    float v = 0.0f;
    if (j < 128)      v = W0[n * 259 + 3 + j];           // node_i block
    else if (j < 256) v = W0[n * 259 + 131 + (j - 128)]; // node_j block
    else if (j < 259) v = W0[n * 259 + (j - 256)];       // edge_attr block
    W0b[t] = (__bf16)v;
  } else if (t < N0 + N2) {
    const int u = t - N0;
    W2b[u] = (__bf16)W2[u];
  } else if (t < N0 + N2 + N3) {
    const int u = t - N0 - N2;
    W3b[u] = (__bf16)W3[u];
  }
}

// x fp32 -> bf16 (one-time; lets the gather be a pure global->LDS bf16 copy)
__global__ __launch_bounds__(256) void edgenet_prep_x(
    const float* __restrict__ x, __bf16* __restrict__ xb, long long n) {
  const long long t = (long long)blockIdx.x * 256 + threadIdx.x;
  if (t < n) xb[t] = (__bf16)x[t];
}

// ---------------------------------------------------------------------------
// NT N-tiles x 4 M-tiles of 16x16, K = NCHUNK*32. A in LDS (row-major bf16),
// B = weight rows (W[n][k], bf16) streamed from global/L2. B loaded once per
// (kc, nt) and reused for all 4 M-tiles; A loaded once per (kc, m) and reused
// for all NT N-tiles.
// A lane layout (16b A 16x32): lane<16 -> K 0..7 & 16..23, lane>=16 -> 8..15 & 24..31.
// B lane layout (16b B 32x16): lane<16 -> K 0..7 & 16..23, lane>=16 -> 8..15 & 24..31, N = lane%16.
// ---------------------------------------------------------------------------
template <int NCHUNK, int NT>
__device__ __forceinline__ void gemm_block(const __bf16* __restrict__ A, int lda,
                                           const __bf16* __restrict__ W, int ldw,
                                           int n0, int nstep, int lane,
                                           v8f acc[NT][MT]) {
  const int row = lane & 15;
  const int hi  = lane >> 4;
#pragma unroll
  for (int nt = 0; nt < NT; ++nt)
#pragma unroll
    for (int m = 0; m < MT; ++m) acc[nt][m] = {};

#pragma unroll
  for (int kc = 0; kc < NCHUNK; ++kc) {
    V16BF a[MT];
    const int ka = kc * 32 + hi * 8;
#pragma unroll
    for (int m = 0; m < MT; ++m) {
      const __bf16* arow = A + (m * 16 + row) * lda;
      a[m].h[0] = *(const v8bf*)(arow + ka);
      a[m].h[1] = *(const v8bf*)(arow + ka + 16);
    }
    const int kb = kc * 32 + hi * 8;
#pragma unroll
    for (int nt = 0; nt < NT; ++nt) {
      const __bf16* brow = W + (size_t)(n0 + nt * nstep + row) * (size_t)ldw;
      V16BF b;
      b.h[0] = *(const v8bf*)(brow + kb);
      b.h[1] = *(const v8bf*)(brow + kb + 16);
#pragma unroll
      for (int m = 0; m < MT; ++m)
        acc[nt][m] = __builtin_amdgcn_wmma_f32_16x16x32_bf16(
            /*neg_a=*/false, a[m].v, /*neg_b=*/false, b.v,
            /*c_mod=*/(short)0, acc[nt][m], /*reuse_a=*/false, /*reuse_b=*/false);
    }
  }
}

// Bias + ELU on f32 accumulator, re-quantize to bf16 into LDS.
// C/D layout: VGPR r, lane -> M = mtile*16 + r + 8*(lane>>4), N = n0 + (lane&15).
__device__ __forceinline__ void elu_store(v8f acc, const float* __restrict__ bias,
                                          int n0, int mtile,
                                          __bf16* __restrict__ H, int ldh, int lane) {
  const int col = n0 + (lane & 15);
  const int hi  = lane >> 4;
  const float bv = bias[col];
#pragma unroll
  for (int r = 0; r < 8; ++r) {
    float v = acc[r] + bv;
    v = (v > 0.0f) ? v : (__expf(v) - 1.0f);
    H[(mtile * 16 + r + 8 * hi) * ldh + col] = (__bf16)v;
  }
}

// ---------------------------------------------------------------------------
// Main: one block = 64 edges, 256 threads = 8 waves.
// ---------------------------------------------------------------------------
__global__ __launch_bounds__(256) void edgenet_main(
    const __bf16* __restrict__ xb, const long long* __restrict__ eidx,
    const float* __restrict__ pos,
    const float* __restrict__ b0, const float* __restrict__ b2,
    const float* __restrict__ b3,
    const __bf16* __restrict__ W0b, const __bf16* __restrict__ W2b,
    const __bf16* __restrict__ W3b,
    float* __restrict__ out, long long E) {
  __shared__ __attribute__((aligned(16))) __bf16 sA[MBLK * KPAD0]; // input / h1
  __shared__ __attribute__((aligned(16))) __bf16 sH[MBLK * LDH];   // h0

  const int t    = threadIdx.x;
  const int lane = t & 31;
  const int w    = t >> 5;                  // wave id 0..7
  const long long e0 = (long long)blockIdx.x * MBLK;

  // ---- Gather: async global->LDS bf16 copy; 16 threads x 32B per edge row ----
  {
    const int sub = t & 15;
#pragma unroll
    for (int rr = 0; rr < MT; ++rr) {
      const int m = (t >> 4) + 16 * rr;     // edge row 0..63
      const long long e = e0 + m;
      __bf16* dstp = sA + m * KPAD0 + sub * 16;   // 16 halves = 32B per thread
      if (e < E) {
        const long long s = eidx[e];
        const long long d = eidx[E + e];
        const __bf16* grow = (sub < 8) ? (xb + s * D_FEAT + sub * 16)
                                       : (xb + d * D_FEAT + (sub - 8) * 16);
#if HAVE_ASYNC_LDS
        __builtin_amdgcn_global_load_async_to_lds_b128(
            (Gv4i*)grow, (Lv4i*)dstp, 0, 0);
        __builtin_amdgcn_global_load_async_to_lds_b128(
            (Gv4i*)(grow + 8), (Lv4i*)(dstp + 8), 0, 0);
#else
        *(v8bf*)dstp       = *(const v8bf*)grow;
        *(v8bf*)(dstp + 8) = *(const v8bf*)(grow + 8);
#endif
        if (sub == 0) {  // edge_attr (K' 256..258) + zero pad (259..287)
          __bf16* tp = sA + m * KPAD0 + 256;
          tp[0] = (__bf16)(pos[d * 3 + 0] - pos[s * 3 + 0]);
          tp[1] = (__bf16)(pos[d * 3 + 1] - pos[s * 3 + 1]);
          tp[2] = (__bf16)(pos[d * 3 + 2] - pos[s * 3 + 2]);
#pragma unroll
          for (int j = 3; j < 32; ++j) tp[j] = (__bf16)0.0f;
        }
      } else {
#pragma unroll
        for (int j = 0; j < 16; ++j) dstp[j] = (__bf16)0.0f;
        if (sub == 0) {
          __bf16* tp = sA + m * KPAD0 + 256;
#pragma unroll
          for (int j = 0; j < 32; ++j) tp[j] = (__bf16)0.0f;
        }
      }
    }
  }
#if HAVE_ASYNC_LDS
#if __has_builtin(__builtin_amdgcn_s_wait_asynccnt)
  __builtin_amdgcn_s_wait_asynccnt(0);
#else
  asm volatile("s_wait_asynccnt 0" ::: "memory");
#endif
#endif
  __syncthreads();

  // ---- Layer 0: [64,288] x W0b[256,288]^T -> h0 [64,256]; 2 N-tiles/wave ----
  {
    v8f acc[2][MT];
    gemm_block<KPAD0 / 32, 2>(sA, KPAD0, W0b, KPAD0, w * 16, 128, lane, acc);
#pragma unroll
    for (int m = 0; m < MT; ++m) {
      elu_store(acc[0][m], b0, w * 16, m, sH, LDH, lane);
      elu_store(acc[1][m], b0, w * 16 + 128, m, sH, LDH, lane);
    }
  }
  __syncthreads();   // h0 ready; sA fully consumed

  // ---- Layer 1: h0 [64,256] x W2b[256,256]^T -> h1 [64,256] (into sA) ----
  {
    v8f acc[2][MT];
    gemm_block<HID / 32, 2>(sH, LDH, W2b, HID, w * 16, 128, lane, acc);
#pragma unroll
    for (int m = 0; m < MT; ++m) {
      elu_store(acc[0][m], b2, w * 16, m, sA, LDH, lane);
      elu_store(acc[1][m], b2, w * 16 + 128, m, sA, LDH, lane);
    }
  }
  __syncthreads();   // h1 ready

  // ---- Layer 2: h1 [64,256] x W3b[128,256]^T -> out [64,128]; 1 N-tile/wave ----
  {
    v8f acc[1][MT];
    gemm_block<HID / 32, 1>(sA, LDH, W3b, HID, w * 16, 0, lane, acc);
    const int col = w * 16 + (lane & 15);
    const int hi  = lane >> 4;
    const float bv = b3[col];
#pragma unroll
    for (int m = 0; m < MT; ++m) {
#pragma unroll
      for (int r = 0; r < 8; ++r) {
        const long long eo = e0 + m * 16 + r + 8 * hi;
        if (eo < E) out[eo * OUTC + col] = acc[0][m][r] + bv;
      }
    }
  }
}

// ---------------------------------------------------------------------------
// Host launcher. d_in order: x, edge_index(int64), pos, W0, b0, W2, b2, W3, b3.
// Workspace (bf16): W0b[256*288] | W2b[256*256] | W3b[128*256] | xb[N*128]
//                   = 344064 B + 2*|x| bytes.
// ---------------------------------------------------------------------------
extern "C" void kernel_launch(void* const* d_in, const int* in_sizes, int n_in,
                              void* d_out, int out_size, void* d_ws, size_t ws_size,
                              hipStream_t stream) {
  const float*     x    = (const float*)d_in[0];
  const long long* eidx = (const long long*)d_in[1];
  const float*     pos  = (const float*)d_in[2];
  const float*     W0   = (const float*)d_in[3];
  const float*     b0   = (const float*)d_in[4];
  const float*     W2   = (const float*)d_in[5];
  const float*     b2   = (const float*)d_in[6];
  const float*     W3   = (const float*)d_in[7];
  const float*     b3   = (const float*)d_in[8];
  float* out = (float*)d_out;

  const long long E     = (long long)in_sizes[1] / 2;
  const long long xelem = (long long)in_sizes[0];     // N_nodes * 128

  __bf16* W0b = (__bf16*)d_ws;
  __bf16* W2b = W0b + (size_t)HID * KPAD0;
  __bf16* W3b = W2b + (size_t)HID * HID;
  __bf16* xb  = W3b + (size_t)OUTC * HID;

  const int prep_elems = HID * KPAD0 + HID * HID + OUTC * HID; // 172032
  edgenet_prep<<<(prep_elems + 255) / 256, 256, 0, stream>>>(W0, W2, W3, W0b, W2b, W3b);
  edgenet_prep_x<<<(int)((xelem + 255) / 256), 256, 0, stream>>>(x, xb, xelem);

  const int nblk = (int)((E + MBLK - 1) / MBLK);
  edgenet_main<<<nblk, 256, 0, stream>>>(xb, eidx, pos, b0, b2, b3,
                                         W0b, W2b, W3b, out, E);
}
